// ResamplerBlock_2482491097098
// MI455X (gfx1250) — compile-verified, matrix operand order checked by reading, not run on hardware
//
#include <hip/hip_runtime.h>
#include <math.h>

typedef __attribute__((ext_vector_type(16))) __bf16 v16bf;
typedef __attribute__((ext_vector_type(8)))  float  v8f;

static __device__ __forceinline__ unsigned short to_bf16(float f) {
  unsigned u = __float_as_uint(f);
  unsigned r = u + 0x7FFFu + ((u >> 16) & 1u);
  return (unsigned short)(r >> 16);
}

static __device__ __forceinline__ v8f wmma_bf16(v16bf a, v16bf b, v8f c) {
  return __builtin_amdgcn_wmma_f32_16x16x32_bf16(false, a, false, b, (short)0, c, false, false);
}

// ---------------------------------------------------------------------------
// Per-row mean / rstd (layernorm stats)
// ---------------------------------------------------------------------------
__global__ __launch_bounds__(256) void row_stats(const float* __restrict__ x,
                                                 float* __restrict__ mean,
                                                 float* __restrict__ rstd, int W) {
  __shared__ float s1[256], s2[256];
  const int r = blockIdx.x;
  const float* p = x + (size_t)r * W;
  float a = 0.f, b = 0.f;
  for (int i = threadIdx.x; i < W; i += 256) { float v = p[i]; a += v; b += v * v; }
  s1[threadIdx.x] = a; s2[threadIdx.x] = b;
  __syncthreads();
  for (int s = 128; s > 0; s >>= 1) {
    if (threadIdx.x < s) { s1[threadIdx.x] += s1[threadIdx.x + s]; s2[threadIdx.x] += s2[threadIdx.x + s]; }
    __syncthreads();
  }
  if (threadIdx.x == 0) {
    float m = s1[0] / W;
    float var = s2[0] / W - m * m;
    mean[r] = m;
    rstd[r] = rsqrtf(var + 1e-5f);
  }
}

// ---------------------------------------------------------------------------
// W_fused_t[n][k] = sum_j W_bottle[k][j] * Wkv[j][n]   (bf16, transposed)
// ---------------------------------------------------------------------------
__global__ __launch_bounds__(256) void fuse_wkv(const float* __restrict__ Wb,
                                                const float* __restrict__ Wkv,
                                                unsigned short* __restrict__ out) {
  int t = blockIdx.x * 256 + threadIdx.x;
  if (t >= 128 * 1024) return;
  int n = t >> 10, k = t & 1023;
  float s = 0.f;
  for (int j = 0; j < 768; j++) s += Wb[(size_t)k * 768 + j] * Wkv[(size_t)j * 128 + n];
  out[t] = to_bf16(s);
}

// f32 [K,N] row-major  ->  bf16 [N,K] (transposed)
__global__ __launch_bounds__(256) void conv_t(const float* __restrict__ in,
                                              unsigned short* __restrict__ out,
                                              int K, int N) {
  int t = blockIdx.x * 256 + threadIdx.x;
  if (t >= K * N) return;
  int n = t / K, k = t % K;
  out[t] = to_bf16(in[(size_t)k * N + n]);
}

// h = hidden + sigmoid(gate) * LN(ao)
__global__ __launch_bounds__(256) void apply_ln_gate(const float* __restrict__ hidden,
                                                     const float* __restrict__ ao,
                                                     const float* __restrict__ mean,
                                                     const float* __restrict__ rstd,
                                                     const float* __restrict__ g,
                                                     const float* __restrict__ bb,
                                                     const float* __restrict__ gate_p,
                                                     float* __restrict__ h) {
  int i = blockIdx.x * 256 + threadIdx.x;
  int row = i / 768, col = i % 768;
  float gate = 1.f / (1.f + __expf(-gate_p[0]));
  h[i] = hidden[i] + gate * ((ao[i] - mean[row]) * rstd[row] * g[col] + bb[col]);
}

// out_bf16 = LN(x)
__global__ __launch_bounds__(256) void ln_to_bf16(const float* __restrict__ x,
                                                  const float* __restrict__ mean,
                                                  const float* __restrict__ rstd,
                                                  const float* __restrict__ g,
                                                  const float* __restrict__ bb,
                                                  unsigned short* __restrict__ out, int W) {
  int i = blockIdx.x * 256 + threadIdx.x;
  int row = i / W, col = i % W;
  out[i] = to_bf16((x[i] - mean[row]) * rstd[row] * g[col] + bb[col]);
}

// ---------------------------------------------------------------------------
// Generic WMMA GEMM with double-buffered LDS ping-pong:
//   C[M,N] = epi(A[M,K] @ B[K,N] + bias)
// Block: 256 thr (8 waves), tile 128x128, K-step 32.
// Next tile's global loads are staged in VGPRs under the current tile's WMMAs.
// Bt stored transposed [N][K] bf16 so all fragment loads are contiguous.
// AMODE 0: A is bf16 [M,K]; AMODE 1: A is f32 with fused layernorm.
// ---------------------------------------------------------------------------
enum { EPI_GELU = 1, EPI_F32 = 2, EPI_RES = 3, EPI_KV = 4, EPI_Q = 5 };

template <int AMODE, int EPI>
__global__ __launch_bounds__(256) void gemm_wmma(
    const void* __restrict__ Aptr, const unsigned short* __restrict__ Bt,
    const float* __restrict__ bias, void* __restrict__ out0, void* __restrict__ out1,
    const float* __restrict__ resid, const float* __restrict__ lnG,
    const float* __restrict__ lnB, const float* __restrict__ mean,
    const float* __restrict__ rstd, int M, int N, int K, float scale) {
  __shared__ unsigned short atile[2][128 * 32];
  __shared__ unsigned short btile[2][128 * 32];
  const int tid = threadIdx.x;
  const int wave = tid >> 5, lane = tid & 31, half = lane >> 4, lr = lane & 15;
  const int m0 = blockIdx.x * 128, n0 = blockIdx.y * 128;
  const int row = tid >> 1, seg = tid & 1;
  const int nsteps = K >> 5;
  v8f acc[8] = {};

  float muA = 0.f, rsA = 0.f;
  if (AMODE == 1) { muA = mean[m0 + row]; rsA = rstd[m0 + row]; }

  uint4 arg0, arg1, brg0, brg1;
  float4 afg[4];

  auto load_regs = [&](int k0) {
    if (AMODE == 0) {
      const uint4* s = (const uint4*)((const unsigned short*)Aptr + (size_t)(m0 + row) * K + k0 + seg * 16);
      arg0 = s[0]; arg1 = s[1];
    } else {
      const float4* s = (const float4*)((const float*)Aptr + (size_t)(m0 + row) * K + k0 + seg * 16);
      afg[0] = s[0]; afg[1] = s[1]; afg[2] = s[2]; afg[3] = s[3];
    }
    const uint4* sb = (const uint4*)(Bt + (size_t)(n0 + row) * K + k0 + seg * 16);
    brg0 = sb[0]; brg1 = sb[1];
  };
  auto store_lds = [&](int buf, int k0) {
    if (AMODE == 0) {
      uint4* d = (uint4*)(&atile[buf][row * 32 + seg * 16]);
      d[0] = arg0; d[1] = arg1;
    } else {
      const float* vv = (const float*)afg;
#pragma unroll
      for (int i = 0; i < 16; i++) {
        int c = k0 + seg * 16 + i;
        atile[buf][row * 32 + seg * 16 + i] = to_bf16((vv[i] - muA) * rsA * lnG[c] + lnB[c]);
      }
    }
    uint4* d = (uint4*)(&btile[buf][row * 32 + seg * 16]);
    d[0] = brg0; d[1] = brg1;
  };

  load_regs(0);
  store_lds(0, 0);
  __syncthreads();

  for (int i = 0; i < nsteps; i++) {
    const int cur = i & 1;
    const bool more = (i + 1) < nsteps;
    if (more) load_regs((i + 1) * 32);   // global loads in flight under the WMMAs
    v16bf af = *(const v16bf*)(&atile[cur][(wave * 16 + lr) * 32 + half * 16]);
#pragma unroll
    for (int sub = 0; sub < 8; sub++) {
      v16bf bf = *(const v16bf*)(&btile[cur][(sub * 16 + lr) * 32 + half * 16]);
      acc[sub] = wmma_bf16(af, bf, acc[sub]);
    }
    if (more) store_lds(1 ^ cur, (i + 1) * 32);
    __syncthreads();
  }

#pragma unroll
  for (int sub = 0; sub < 8; sub++) {
    int col = n0 + sub * 16 + lr;
    float bv = bias[col];
#pragma unroll
    for (int v = 0; v < 8; v++) {
      int r = m0 + wave * 16 + v + 8 * half;
      float val = (acc[sub][v] + bv) * scale;
      if (EPI == EPI_GELU) {
        val = 0.5f * val * (1.f + erff(val * 0.70710678118f));
        ((unsigned short*)out0)[(size_t)r * N + col] = to_bf16(val);
      } else if (EPI == EPI_F32) {
        ((float*)out0)[(size_t)r * N + col] = val;
      } else if (EPI == EPI_RES) {
        ((float*)out0)[(size_t)r * N + col] = val + resid[(size_t)r * N + col];
      } else if (EPI == EPI_KV) {
        if (col < 64) ((unsigned short*)out0)[(size_t)r * 64 + col] = to_bf16(val);
        else          ((unsigned short*)out1)[(size_t)r * 64 + (col - 64)] = to_bf16(val);
      } else if (EPI == EPI_Q) {
        int b = r >> 8, q = r & 255, hh = col >> 6, d = col & 63;
        ((unsigned short*)out0)[((size_t)(b * 12 + hh) * 256 + q) * 64 + d] = to_bf16(val);
      }
    }
  }
}

// ---------------------------------------------------------------------------
// Flash attention: grid = B*NH*2, block = 256 (8 waves x 16 query rows).
// Streams 64-key chunks with double-buffered K/V tiles; next chunk's global
// loads are staged in VGPRs under the current chunk's WMMAs + softmax VALU.
// ---------------------------------------------------------------------------
__global__ __launch_bounds__(256) void attn_kernel(const unsigned short* __restrict__ qws,
                                                   const unsigned short* __restrict__ Kws,
                                                   const unsigned short* __restrict__ Vws,
                                                   unsigned short* __restrict__ attn_out) {
  __shared__ unsigned short ktile[2][64 * 64];
  __shared__ unsigned short vtile[2][64 * 64];   // transposed: [d][key]
  __shared__ unsigned short pscr[8 * 16 * 64];
  const int tid = threadIdx.x, wave = tid >> 5, lane = tid & 31;
  const int half = lane >> 4, lr = lane & 15;
  const int qt = blockIdx.x & 1, bh = blockIdx.x >> 1;
  const int b = bh / 12, h = bh % 12;
  const int qrow0 = qt * 128 + wave * 16;

  const size_t qbase = ((size_t)(bh * 256 + qrow0 + lr)) * 64;
  v16bf aq0 = *(const v16bf*)(qws + qbase + half * 16);
  v16bf aq1 = *(const v16bf*)(qws + qbase + 32 + half * 16);

  v8f o[4] = {};
  float mrow[8], lrow[8];
#pragma unroll
  for (int v = 0; v < 8; v++) { mrow[v] = -1e30f; lrow[v] = 0.f; }

  const int key_t = tid >> 2, seg = tid & 3;
  uint4 krg0, krg1, vrg0, vrg1;

  auto load_regs = [&](int kc) {
    size_t gbase = ((size_t)(b * 4096 + kc * 64 + key_t)) * 64 + seg * 16;
    const uint4* sk = (const uint4*)(Kws + gbase);
    krg0 = sk[0]; krg1 = sk[1];
    const uint4* sv = (const uint4*)(Vws + gbase);
    vrg0 = sv[0]; vrg1 = sv[1];
  };
  auto store_lds = [&](int buf) {
    uint4* d = (uint4*)(&ktile[buf][key_t * 64 + seg * 16]);
    d[0] = krg0; d[1] = krg1;
    union { uint4 u[2]; unsigned short s[16]; } vv;
    vv.u[0] = vrg0; vv.u[1] = vrg1;
#pragma unroll
    for (int i = 0; i < 16; i++) vtile[buf][(seg * 16 + i) * 64 + key_t] = vv.s[i];
  };

  load_regs(0);
  store_lds(0);
  __syncthreads();

  for (int kc = 0; kc < 64; kc++) {
    const int cur = kc & 1;
    const bool more = (kc + 1) < 64;
    if (more) load_regs(kc + 1);     // next chunk's loads fly under this chunk's math

    v8f s4[4];
#pragma unroll
    for (int sub = 0; sub < 4; sub++) {
      v8f z = {};
      v16bf bk0 = *(const v16bf*)(&ktile[cur][(sub * 16 + lr) * 64 + half * 16]);
      v16bf bk1 = *(const v16bf*)(&ktile[cur][(sub * 16 + lr) * 64 + 32 + half * 16]);
      z = wmma_bf16(aq0, bk0, z);
      z = wmma_bf16(aq1, bk1, z);
      s4[sub] = z;
    }
#pragma unroll
    for (int v = 0; v < 8; v++) {
      float rm = fmaxf(fmaxf(s4[0][v], s4[1][v]), fmaxf(s4[2][v], s4[3][v]));
#pragma unroll
      for (int off = 1; off < 16; off <<= 1) rm = fmaxf(rm, __shfl_xor(rm, off, 32));
      float nm = fmaxf(mrow[v], rm);
      float corr = __expf(mrow[v] - nm);
      float rs = 0.f;
#pragma unroll
      for (int sub = 0; sub < 4; sub++) { float p = __expf(s4[sub][v] - nm); s4[sub][v] = p; rs += p; }
#pragma unroll
      for (int off = 1; off < 16; off <<= 1) rs += __shfl_xor(rs, off, 32);
      lrow[v] = lrow[v] * corr + rs;
#pragma unroll
      for (int sub = 0; sub < 4; sub++) o[sub][v] *= corr;
      mrow[v] = nm;
    }
#pragma unroll
    for (int sub = 0; sub < 4; sub++)
#pragma unroll
      for (int v = 0; v < 8; v++)
        pscr[wave * 1024 + (v + 8 * half) * 64 + sub * 16 + lr] = to_bf16(s4[sub][v]);

    if (more) store_lds(1 ^ cur);

    __syncthreads();   // publishes pscr; also orders next-buffer stores

    v16bf p0 = *(const v16bf*)(pscr + wave * 1024 + lr * 64 + half * 16);
    v16bf p1 = *(const v16bf*)(pscr + wave * 1024 + lr * 64 + 32 + half * 16);
#pragma unroll
    for (int sub = 0; sub < 4; sub++) {
      v16bf bv0 = *(const v16bf*)(&vtile[cur][(sub * 16 + lr) * 64 + half * 16]);
      v16bf bv1 = *(const v16bf*)(&vtile[cur][(sub * 16 + lr) * 64 + 32 + half * 16]);
      o[sub] = wmma_bf16(p0, bv0, o[sub]);
      o[sub] = wmma_bf16(p1, bv1, o[sub]);
    }
    __syncthreads();   // all reads of buf[cur] done before it is overwritten at kc+1
  }

#pragma unroll
  for (int sub = 0; sub < 4; sub++)
#pragma unroll
    for (int v = 0; v < 8; v++) {
      int r = b * 256 + qt * 128 + wave * 16 + v + 8 * half;
      int col = h * 64 + sub * 16 + lr;
      attn_out[(size_t)r * 768 + col] = to_bf16(o[sub][v] / lrow[v]);
    }
}

// ---------------------------------------------------------------------------
extern "C" void kernel_launch(void* const* d_in, const int* in_sizes, int n_in,
                              void* d_out, int out_size, void* d_ws, size_t ws_size,
                              hipStream_t stream) {
  const float* hidden   = (const float*)d_in[0];
  const float* img      = (const float*)d_in[1];
  const float* ln_img_g = (const float*)d_in[2];
  const float* ln_img_b = (const float*)d_in[3];
  const float* W_bottle = (const float*)d_in[4];
  const float* ln_h_g   = (const float*)d_in[5];
  const float* ln_h_b   = (const float*)d_in[6];
  const float* Wq       = (const float*)d_in[7];
  const float* bq       = (const float*)d_in[8];
  const float* Wkv      = (const float*)d_in[9];
  const float* bkv      = (const float*)d_in[10];
  const float* Wo       = (const float*)d_in[11];
  const float* bo       = (const float*)d_in[12];
  const float* ln_ao_g  = (const float*)d_in[13];
  const float* ln_ao_b  = (const float*)d_in[14];
  const float* gate_p   = (const float*)d_in[15];
  const float* ln_f_g   = (const float*)d_in[16];
  const float* ln_f_b   = (const float*)d_in[17];
  const float* W1       = (const float*)d_in[18];
  const float* b1       = (const float*)d_in[19];
  const float* W2       = (const float*)d_in[20];
  const float* b2       = (const float*)d_in[21];

  char* base = (char*)d_ws;
  size_t off = 0;
  auto alloc = [&](size_t bytes) -> char* {
    char* p = base + off;
    off += (bytes + 255) & ~(size_t)255;
    return p;
  };

  float* imean  = (float*)alloc(32768 * 4);
  float* irstd  = (float*)alloc(32768 * 4);
  float* hmean  = (float*)alloc(2048 * 4);
  float* hrstd  = (float*)alloc(2048 * 4);
  float* aomean = (float*)alloc(2048 * 4);
  float* aorstd = (float*)alloc(2048 * 4);
  float* fmean  = (float*)alloc(2048 * 4);
  float* frstd  = (float*)alloc(2048 * 4);
  unsigned short* wkvT  = (unsigned short*)alloc((size_t)128 * 1024 * 2);
  unsigned short* wqT   = (unsigned short*)alloc((size_t)768 * 768 * 2);
  unsigned short* woT   = (unsigned short*)alloc((size_t)768 * 768 * 2);
  unsigned short* w1T   = (unsigned short*)alloc((size_t)3072 * 768 * 2);
  unsigned short* w2T   = (unsigned short*)alloc((size_t)768 * 3072 * 2);
  unsigned short* Kws   = (unsigned short*)alloc((size_t)32768 * 64 * 2);
  unsigned short* Vws   = (unsigned short*)alloc((size_t)32768 * 64 * 2);
  unsigned short* qws   = (unsigned short*)alloc((size_t)96 * 256 * 64 * 2);
  unsigned short* attnw = (unsigned short*)alloc((size_t)2048 * 768 * 2);
  float*          ao    = (float*)alloc((size_t)2048 * 768 * 4);
  float*          hbuf  = (float*)alloc((size_t)2048 * 768 * 4);
  unsigned short* ffnin = (unsigned short*)alloc((size_t)2048 * 768 * 2);
  unsigned short* act   = (unsigned short*)alloc((size_t)2048 * 3072 * 2);

  // 1. layernorm stats
  row_stats<<<32768, 256, 0, stream>>>(img, imean, irstd, 1024);
  row_stats<<<2048, 256, 0, stream>>>(hidden, hmean, hrstd, 768);

  // 2. weight prep: fold W_bottle into Wkv; transpose+convert the rest
  fuse_wkv<<<(128 * 1024) / 256, 256, 0, stream>>>(W_bottle, Wkv, wkvT);
  conv_t<<<(768 * 768) / 256, 256, 0, stream>>>(Wq, wqT, 768, 768);
  conv_t<<<(768 * 768) / 256, 256, 0, stream>>>(Wo, woT, 768, 768);
  conv_t<<<(768 * 3072) / 256, 256, 0, stream>>>(W1, w1T, 768, 3072);
  conv_t<<<(3072 * 768) / 256, 256, 0, stream>>>(W2, w2T, 3072, 768);

  // 3. KV = LN(img) @ W_fused + bkv   (fused LN in A-loader)
  gemm_wmma<1, EPI_KV><<<dim3(256, 1), 256, 0, stream>>>(
      img, wkvT, bkv, Kws, Vws, nullptr, ln_img_g, ln_img_b, imean, irstd,
      32768, 128, 1024, 1.0f);

  // 4. Q = (LN(hidden) @ Wq + bq) / sqrt(64), head-major layout
  gemm_wmma<1, EPI_Q><<<dim3(16, 6), 256, 0, stream>>>(
      hidden, wqT, bq, qws, nullptr, nullptr, ln_h_g, ln_h_b, hmean, hrstd,
      2048, 768, 768, 0.125f);

  // 5. flash attention
  attn_kernel<<<192, 256, 0, stream>>>(qws, Kws, Vws, attnw);

  // 6. out @ Wo + bo
  gemm_wmma<0, EPI_F32><<<dim3(16, 6), 256, 0, stream>>>(
      attnw, woT, bo, ao, nullptr, nullptr, nullptr, nullptr, nullptr, nullptr,
      2048, 768, 768, 1.0f);

  // 7. h = hidden + sigmoid(gate) * LN(ao)
  row_stats<<<2048, 256, 0, stream>>>(ao, aomean, aorstd, 768);
  apply_ln_gate<<<6144, 256, 0, stream>>>(hidden, ao, aomean, aorstd,
                                          ln_ao_g, ln_ao_b, gate_p, hbuf);

  // 8. ffn_in = LN(h)
  row_stats<<<2048, 256, 0, stream>>>(hbuf, fmean, frstd, 768);
  ln_to_bf16<<<6144, 256, 0, stream>>>(hbuf, fmean, frstd, ln_f_g, ln_f_b, ffnin, 768);

  // 9. FFN
  gemm_wmma<0, EPI_GELU><<<dim3(16, 24), 256, 0, stream>>>(
      ffnin, w1T, b1, act, nullptr, nullptr, nullptr, nullptr, nullptr, nullptr,
      2048, 3072, 768, 1.0f);
  gemm_wmma<0, EPI_RES><<<dim3(16, 6), 256, 0, stream>>>(
      act, w2T, b2, d_out, nullptr, hbuf, nullptr, nullptr, nullptr, nullptr,
      2048, 768, 3072, 1.0f);

  (void)in_sizes; (void)n_in; (void)out_size; (void)ws_size;
}
